// TemporalAttention_87033217286265
// MI455X (gfx1250) — matrix operand order repaired and checked by reading, not compiled, not run on hardware
//
#include <hip/hip_runtime.h>

// ---------------------------------------------------------------- constants
constexpr int B_ = 4;
constexpr int C_ = 256;
constexpr int L_ = 4096;
constexpr int K_ = 32;     // key channels = C/8; == WMMA bf16 K-step

constexpr int WAVES_ = 4;              // waves per flash workgroup
constexpr int ROWS_  = WAVES_ * 16;    // query rows per workgroup (64)
constexpr int CHUNK_ = 32;             // keys per inner chunk

// ---------------------------------------------------------------- types
typedef __attribute__((ext_vector_type(16))) __bf16 v16bf;
typedef __attribute__((ext_vector_type(8)))  float  v8f;
typedef __attribute__((ext_vector_type(4)))  int    v4i;
typedef unsigned int u32x4 __attribute__((ext_vector_type(4)));
typedef int          i32x4 __attribute__((ext_vector_type(4)));
typedef int          i32x8 __attribute__((ext_vector_type(8)));

#if __has_builtin(__builtin_amdgcn_tensor_load_to_lds)
#define HAVE_TDM 1
#else
#define HAVE_TDM 0
#endif

// ---------------------------------------------------------------- helpers
__device__ __forceinline__ v8f wmma_bf16(v16bf a, v16bf b, v8f c) {
  // D = A(16x32 bf16) * B(32x16 bf16) + C(16x16 f32)
  return __builtin_amdgcn_wmma_f32_16x16x32_bf16(
      /*neg_a=*/false, a, /*neg_b=*/false, b,
      /*c_mod=*/(short)0, c, /*reuse_a=*/false, /*reuse_b=*/false);
}

// A-matrix 16x32 bf16 tile from a row-major [16 x >=32] source (ld elems/row).
__device__ __forceinline__ v16bf load_a16x32(const __bf16* src, int ld, int lane) {
  const int row  = lane & 15;
  const int half = lane >> 4;
  const __bf16* p = src + (size_t)row * ld + half * 8;
  v16bf r;
#pragma unroll
  for (int i = 0; i < 8; ++i) { r[i] = p[i]; r[8 + i] = p[16 + i]; }
  return r;
}

// B-matrix 32x16 bf16 tile where each B-column n is contiguous at src + n*ld.
__device__ __forceinline__ v16bf load_b32x16(const __bf16* src, size_t ld, int lane) {
  const int n    = lane & 15;
  const int half = lane >> 4;
  const __bf16* p = src + (size_t)n * ld + half * 8;
  v16bf r;
#pragma unroll
  for (int i = 0; i < 8; ++i) { r[i] = p[i]; r[8 + i] = p[16 + i]; }
  return r;
}

#if HAVE_TDM
// TDM: DMA a 2-D bf16 tile (tile_d1 rows x tile_d0 elems, row stride stride0
// elems) from global memory into contiguous LDS. D# per ISA 08_async_tensor.
__device__ __forceinline__ void tdm_load_2d(unsigned lds_addr, const void* gaddr,
                                            unsigned tensor_d0, unsigned tensor_d1,
                                            unsigned tile_d0, unsigned tile_d1,
                                            unsigned long long stride0) {
  const unsigned long long ga = (unsigned long long)(uintptr_t)gaddr;
  u32x4 g0;
  g0[0] = 1u;                                            // count=1 user D#
  g0[1] = lds_addr;                                      // LDS byte address
  g0[2] = (unsigned)(ga & 0xFFFFFFFFu);                  // global_addr lo
  g0[3] = (unsigned)((ga >> 32) & 0x01FFFFFFu) | (2u << 30); // addr hi | type=2
  i32x8 g1;
  g1[0] = (int)(1u << 16);                               // data_size=1 (2B)
  g1[1] = (int)((tensor_d0 & 0xFFFFu) << 16);            // abar=0 | td0.lo16
  g1[2] = (int)((tensor_d0 >> 16) | ((tensor_d1 & 0xFFFFu) << 16));
  g1[3] = (int)((tensor_d1 >> 16) | (tile_d0 << 16));
  g1[4] = (int)(tile_d1 & 0xFFFFu);                      // tile_d1 | tile_d2=0
  g1[5] = (int)(stride0 & 0xFFFFFFFFull);                // dim0 stride lo32
  g1[6] = (int)((stride0 >> 32) & 0xFFFFull);            // stride hi | s1.lo=0
  g1[7] = 0;
  i32x4 z4 = {0, 0, 0, 0};
#if __clang_major__ >= 23
  i32x8 z8 = {0, 0, 0, 0, 0, 0, 0, 0};
  __builtin_amdgcn_tensor_load_to_lds(g0, g1, z4, z4, z8, 0);
#else
  __builtin_amdgcn_tensor_load_to_lds(g0, g1, z4, z4, 0);
#endif
}
#endif

// 16B global -> LDS copy (fallback staging path when TDM builtin is absent).
__device__ __forceinline__ void copy16_g2l(const __bf16* g, __bf16* l) {
#if __has_builtin(__builtin_amdgcn_global_load_async_to_lds_b128)
  typedef __attribute__((address_space(1))) v4i gv4i;
  typedef __attribute__((address_space(3))) v4i lv4i;
  __builtin_amdgcn_global_load_async_to_lds_b128((gv4i*)g, (lv4i*)l, 0, 0);
#else
  *(v4i*)l = *(const v4i*)g;
#endif
}

__device__ __forceinline__ void wait_async_copies() {
#if __has_builtin(__builtin_amdgcn_global_load_async_to_lds_b128)
#if __has_builtin(__builtin_amdgcn_s_wait_asynccnt)
  __builtin_amdgcn_s_wait_asynccnt(0);
#else
  asm volatile("s_wait_asynccnt 0" ::: "memory");
#endif
#endif
}

// ---------------------------------------------------------------- kernels
// x[B][C][L] f32 -> xT[B][L][C] bf16
__global__ void xpose_bf16_kernel(const float* __restrict__ x, __bf16* __restrict__ xT) {
  size_t idx = (size_t)blockIdx.x * blockDim.x + threadIdx.x;
  if (idx >= (size_t)B_ * C_ * L_) return;
  int l = (int)(idx % L_);
  size_t t = idx / L_;
  int c = (int)(t % C_);
  int b = (int)(t / C_);
  xT[((size_t)b * L_ + l) * C_ + c] = (__bf16)x[idx];
}

__global__ void cvt_bf16_kernel(const float* __restrict__ in, __bf16* __restrict__ out, int n) {
  int i = blockIdx.x * blockDim.x + threadIdx.x;
  if (i < n) out[i] = (__bf16)in[i];
}

// OUT[l,n] = sum_c xT[l,c] * W[n,c] + bias[n]   (one wave -> one 16x16 tile)
__global__ void __launch_bounds__(32)
proj_wmma_kernel(const __bf16* __restrict__ xT, const __bf16* __restrict__ W,
                 const float* __restrict__ bias, __bf16* __restrict__ out,
                 int ldo, int outTransposed) {
  const int lane = threadIdx.x & 31;
  const int l0 = blockIdx.x * 16;
  const int n0 = blockIdx.y * 16;
  const int b  = blockIdx.z;
  const __bf16* A = xT + ((size_t)b * L_ + l0) * C_;

  v8f acc = {};
#pragma unroll
  for (int c0 = 0; c0 < C_; c0 += 32) {
    v16bf a  = load_a16x32(A + c0, C_, lane);
    v16bf bb = load_b32x16(W + (size_t)n0 * C_ + c0, C_, lane);
    acc = wmma_bf16(a, bb, acc);
  }

  const int n     = n0 + (lane & 15);
  const int rbase = l0 + (lane >> 4) * 8;
  const float bv  = bias[n];

  if (!outTransposed) {
    __bf16* o = out + (size_t)b * L_ * ldo;
#pragma unroll
    for (int r = 0; r < 8; ++r) o[(size_t)(rbase + r) * ldo + n] = (__bf16)(acc[r] + bv);
  } else {
    __bf16* o = out + (size_t)b * C_ * L_;
#pragma unroll
    for (int r = 0; r < 8; ++r) o[(size_t)n * L_ + rbase + r] = (__bf16)(acc[r] + bv);
  }
}

// Cooperative fused flash attention + epilogue.
// 4 waves / 64 query rows per block; K/V chunks staged in LDS by the TDM
// (double buffer, TENSORcnt-synchronized), shared by all waves.
__global__ void __launch_bounds__(WAVES_ * 32)
flash_attn_kernel(const __bf16* __restrict__ qb, const __bf16* __restrict__ kb,
                  const __bf16* __restrict__ vT, const float* __restrict__ x,
                  const float* __restrict__ gamma, float* __restrict__ out) {
  const int tid  = threadIdx.x;
  const int lane = tid & 31;
  const int wave = tid >> 5;
  const int b    = blockIdx.y;
  const int l0   = blockIdx.x * ROWS_ + wave * 16;   // this wave's query rows

  const __bf16* q_ = qb + ((size_t)b * L_ + l0) * K_;   // [16][32]
  const __bf16* k_ = kb + (size_t)b * L_ * K_;          // [L][32]
  const __bf16* v_ = vT + (size_t)b * C_ * L_;          // [C][L]

  __shared__ __align__(16) __bf16 kls[2][CHUNK_ * K_];        //  4 KB
  __shared__ __align__(16) __bf16 vls[2][C_ * CHUNK_];        // 32 KB
  __shared__ __align__(16) __bf16 pls[WAVES_][16 * CHUNK_];   //  4 KB

  // ---- stage one 32-key chunk (K tile 32x32, V tile 256x32) into LDS
  auto stage_chunk = [&](int m0, int buf) {
#if HAVE_TDM
    if (wave == 0) {  // TDM issue is per-wave; one wave drives both DMAs
      tdm_load_2d((unsigned)(uintptr_t)&kls[buf][0], k_ + (size_t)m0 * K_,
                  /*tensor_d0=*/K_, /*tensor_d1=*/L_,
                  /*tile_d0=*/K_, /*tile_d1=*/CHUNK_, /*stride0=*/K_);
      tdm_load_2d((unsigned)(uintptr_t)&vls[buf][0], v_ + m0,
                  /*tensor_d0=*/L_, /*tensor_d1=*/C_,
                  /*tile_d0=*/CHUNK_, /*tile_d1=*/C_, /*stride0=*/L_);
    }
#else
    {  // k: 32 rows x 64B -> 128 segments, one per thread
      const int row = tid >> 2, part = tid & 3;
      copy16_g2l(k_ + (size_t)(m0 + row) * K_ + part * 8,
                 &kls[buf][row * K_ + part * 8]);
    }
#pragma unroll
    for (int j = 0; j < 8; ++j) {  // v: 256 ch x 64B -> 1024 segments
      const int s = tid * 8 + j;
      const int c = s >> 2, part = s & 3;
      copy16_g2l(v_ + (size_t)c * L_ + m0 + part * 8,
                 &vls[buf][c * CHUNK_ + part * 8]);
    }
#endif
  };
  auto stage_sync = [&]() {
#if HAVE_TDM
    if (wave == 0) __builtin_amdgcn_s_wait_tensorcnt(0);
#else
    wait_async_copies();
#endif
    __syncthreads();
  };

  const v16bf qa = load_a16x32(q_, K_, lane);
  const v8f zero = {};

  float mrow[8], lrow[8];
#pragma unroll
  for (int r = 0; r < 8; ++r) { mrow[r] = -3.0e38f; lrow[r] = 0.f; }
  v8f O[16];
#pragma unroll
  for (int t = 0; t < 16; ++t) O[t] = zero;

  const int col  = lane & 15;
  const int rofs = (lane >> 4) * 8;

  // prologue: stage chunk 0
  stage_chunk(0, 0);
  stage_sync();

  constexpr int NCHUNK = L_ / CHUNK_;   // 128
  for (int i = 0; i < NCHUNK; ++i) {
    const int cur = i & 1;
    if (i + 1 < NCHUNK) stage_chunk((i + 1) * CHUNK_, cur ^ 1);  // overlap DMA

    // ---- S = q @ k^T for 32 keys (two 16x16 tiles), operands from LDS
    v16bf bk0 = load_b32x16(&kls[cur][0],       K_, lane);
    v16bf bk1 = load_b32x16(&kls[cur][16 * K_], K_, lane);
    v8f s0 = wmma_bf16(qa, bk0, zero);
    v8f s1 = wmma_bf16(qa, bk1, zero);

    // ---- online softmax (row = VGPR r within a 16-lane half)
#pragma unroll
    for (int r = 0; r < 8; ++r) {
      float mx = fmaxf(s0[r], s1[r]);
#pragma unroll
      for (int off = 8; off; off >>= 1) mx = fmaxf(mx, __shfl_xor(mx, off, 16));
      const float mnew = fmaxf(mrow[r], mx);
      const float corr = __expf(mrow[r] - mnew);
      const float p0 = __expf(s0[r] - mnew);
      const float p1 = __expf(s1[r] - mnew);
      float ps = p0 + p1;
#pragma unroll
      for (int off = 8; off; off >>= 1) ps += __shfl_xor(ps, off, 16);
      lrow[r] = lrow[r] * corr + ps;
      mrow[r] = mnew;
#pragma unroll
      for (int t = 0; t < 16; ++t) O[t][r] *= corr;
      pls[wave][(r + rofs) * CHUNK_ + col]      = (__bf16)p0;  // C/D -> LDS
      pls[wave][(r + rofs) * CHUNK_ + col + 16] = (__bf16)p1;
    }
    asm volatile("" ::: "memory");     // per-wave private tile; LDS in-order
    const v16bf pa = load_a16x32(pls[wave], CHUNK_, lane);
    asm volatile("" ::: "memory");

    // ---- O += P @ V  (16 channel tiles of 16, B-tiles from LDS)
#pragma unroll
    for (int t = 0; t < 16; ++t) {
      v16bf bv = load_b32x16(&vls[cur][(t * 16) * CHUNK_], CHUNK_, lane);
      O[t] = wmma_bf16(pa, bv, O[t]);
    }

    stage_sync();   // next-chunk DMA complete + all waves done reading cur
  }

  // ---- epilogue: out[b,c,l] = gamma * (O/l) + x[b,c,l]
  const float g = gamma[0];
  const int rbase = l0 + rofs;
#pragma unroll
  for (int t = 0; t < 16; ++t) {
    const int c = t * 16 + col;
    const float* xr = x   + ((size_t)b * C_ + c) * L_;
    float*      orp = out + ((size_t)b * C_ + c) * L_;
#pragma unroll
    for (int r = 0; r < 8; ++r) {
      const int l = rbase + r;
      orp[l] = g * (O[t][r] / lrow[r]) + xr[l];
    }
  }
}

// ---------------------------------------------------------------- launch
extern "C" void kernel_launch(void* const* d_in, const int* in_sizes, int n_in,
                              void* d_out, int out_size, void* d_ws, size_t ws_size,
                              hipStream_t stream) {
  (void)in_sizes; (void)n_in; (void)out_size; (void)ws_size;
  const float* x     = (const float*)d_in[0];
  const float* Wq    = (const float*)d_in[1];
  const float* bq    = (const float*)d_in[2];
  const float* Wk    = (const float*)d_in[3];
  const float* bk    = (const float*)d_in[4];
  const float* Wv    = (const float*)d_in[5];
  const float* bv    = (const float*)d_in[6];
  const float* gamma = (const float*)d_in[7];

  char* ws = (char*)d_ws;
  __bf16* xT   = (__bf16*)ws; ws += (size_t)B_ * L_ * C_ * 2;   // 8 MB
  __bf16* qb16 = (__bf16*)ws; ws += (size_t)B_ * L_ * K_ * 2;   // 1 MB
  __bf16* kb16 = (__bf16*)ws; ws += (size_t)B_ * L_ * K_ * 2;   // 1 MB
  __bf16* vT16 = (__bf16*)ws; ws += (size_t)B_ * C_ * L_ * 2;   // 8 MB
  __bf16* Wq16 = (__bf16*)ws; ws += (size_t)K_ * C_ * 2;
  __bf16* Wk16 = (__bf16*)ws; ws += (size_t)K_ * C_ * 2;
  __bf16* Wv16 = (__bf16*)ws; ws += (size_t)C_ * C_ * 2;

  // 1) convert / transpose inputs to bf16
  {
    size_t n = (size_t)B_ * C_ * L_;
    xpose_bf16_kernel<<<(unsigned)((n + 255) / 256), 256, 0, stream>>>(x, xT);
    cvt_bf16_kernel<<<(K_ * C_ + 255) / 256, 256, 0, stream>>>(Wq, Wq16, K_ * C_);
    cvt_bf16_kernel<<<(K_ * C_ + 255) / 256, 256, 0, stream>>>(Wk, Wk16, K_ * C_);
    cvt_bf16_kernel<<<(C_ * C_ + 255) / 256, 256, 0, stream>>>(Wv, Wv16, C_ * C_);
  }

  // 2) WMMA projections
  {
    dim3 gqk(L_ / 16, K_ / 16, B_);
    proj_wmma_kernel<<<gqk, 32, 0, stream>>>(xT, Wq16, bq, qb16, K_, 0);
    proj_wmma_kernel<<<gqk, 32, 0, stream>>>(xT, Wk16, bk, kb16, K_, 0);
    dim3 gv(L_ / 16, C_ / 16, B_);
    proj_wmma_kernel<<<gv, 32, 0, stream>>>(xT, Wv16, bv, vT16, 0, 1);
  }

  // 3) cooperative fused flash attention + residual epilogue
  {
    dim3 gf(L_ / ROWS_, B_);
    flash_attn_kernel<<<gf, WAVES_ * 32, 0, stream>>>(qb16, kb16, vT16, x, gamma,
                                                      (float*)d_out);
  }
}
